// MultiScaleAttentionBlock_86758339379249
// MI455X (gfx1250) — compile-verified, matrix operand order checked by reading, not run on hardware
//
#include <hip/hip_runtime.h>
#include <hip/hip_bf16.h>
#include <math.h>

typedef _Float16 half_t;
typedef __attribute__((ext_vector_type(8)))  _Float16 v8h;
typedef __attribute__((ext_vector_type(16))) _Float16 v16h;
typedef __attribute__((ext_vector_type(8)))  float    v8f;

// model constants
#define R0 32768   // 128 windows * 256 tokens
#define R1 2048    // 8 windows * 256 tokens

__device__ __forceinline__ int kmap16(int e, int lane) {
  // ISA 7.12.2: 16-bit A-matrix 16x32 per-lane K mapping (also used for B via W^T)
  return ((e < 8) ? e : (e + 8)) + ((lane & 16) ? 8 : 0);
}

// ---------- Wt[n*K+k] = (f16) W[k*N+n] ----------
__global__ void tcvt_kernel(const float* __restrict__ W, half_t* __restrict__ Wt, int K, int N) {
  int i = blockIdx.x * 256 + threadIdx.x;
  if (i >= K * N) return;
  int n = i / K, k = i - n * K;
  Wt[i] = (half_t)W[(size_t)k * N + n];
}

// ---------- relative position embed table: tab[t][d], t,d in [0,256) ----------
__global__ void pe_table_kernel(const float* __restrict__ w1, const float* __restrict__ b1,
                                const float* __restrict__ w2, float* __restrict__ tab) {
  __shared__ float h[512];
  int t = blockIdx.x, tid = threadIdx.x;
  float c0 = ((float)(t >> 4) - 8.f) / 8.f;
  float c1 = ((float)(t & 15) - 8.f) / 8.f;
  for (int j = tid; j < 512; j += 256) {
    float z = c0 * w1[j] + c1 * w1[512 + j] + b1[j];
    h[j] = z > 0.f ? z : 0.f;
  }
  __syncthreads();
  float acc = 0.f;
  for (int j = 0; j < 512; ++j) acc += h[j] * w2[(size_t)j * 256 + tid];
  tab[(size_t)t * 256 + tid] = acc;
}

// ---------- x0 = scale0 + pe_tab (broadcast over windows) ----------
__global__ void add_pe_kernel(const float* __restrict__ x, const float* __restrict__ tab,
                              float* __restrict__ out) {
  size_t i = (size_t)blockIdx.x * 256 + threadIdx.x;
  out[i] = x[i] + tab[i & (size_t)65535];
}

// ---------- x1 = scale1 + pe_tab1 + bottom_up(x0) ----------
__global__ void bottom_up_kernel(const float* __restrict__ scale1, const float* __restrict__ tab1,
                                 const float* __restrict__ x0, float* __restrict__ x1) {
  int wt = blockIdx.x;                 // 0..8*256-1
  int w1 = wt >> 8, t1 = wt & 255, c = threadIdx.x;
  int b = w1 >> 2, c1y = (w1 >> 1) & 1, c1x = w1 & 1;
  int ty = t1 >> 4, tx = t1 & 15;
  int Y = c1y * 16 + ty, X = c1x * 16 + tx;           // coarse 32x32 grid
  int g0y = Y >> 2, py = Y & 3, g0x = X >> 2, px = X & 3;
  int w0 = b * 64 + g0y * 8 + g0x;
  const float* base = x0 + ((size_t)w0 * 256) * 256 + c;
  float m = -3.4e38f;
  for (int dy = 0; dy < 4; ++dy)
    for (int dx = 0; dx < 4; ++dx) {
      int tok = (py * 4 + dy) * 16 + (px * 4 + dx);
      m = fmaxf(m, base[(size_t)tok * 256]);
    }
  size_t i = ((size_t)w1 * 256 + t1) * 256 + c;
  x1[i] = scale1[i] + tab1[(size_t)t1 * 256 + c] + m;
}

// ---------- LayerNorm -> f16 rows ----------
__global__ void ln_kernel(const float* __restrict__ x, const float* __restrict__ g,
                          const float* __restrict__ b, half_t* __restrict__ out) {
  __shared__ float red[256];
  int row = blockIdx.x, tid = threadIdx.x;
  float v = x[(size_t)row * 256 + tid];
  red[tid] = v;
  __syncthreads();
  for (int s = 128; s > 0; s >>= 1) { if (tid < s) red[tid] += red[tid + s]; __syncthreads(); }
  float mean = red[0] * (1.f / 256.f);
  __syncthreads();
  float d = v - mean;
  red[tid] = d * d;
  __syncthreads();
  for (int s = 128; s > 0; s >>= 1) { if (tid < s) red[tid] += red[tid + s]; __syncthreads(); }
  float rstd = rsqrtf(red[0] * (1.f / 256.f) + 1e-5f);
  out[(size_t)row * 256 + tid] = (half_t)(d * rstd * g[tid] + b[tid]);
}

// ---------- WMMA GEMM: out = epilogue(A(rows x K, f16) * Wt^T + bias) ----------
// mode 0: f32 store, 1: f32 residual add (+=), 2: f16 store, 3: f16 gelu,
// mode 4: KV split store -> K half (col<256) to outp (pitch 512),
//         V half (col>=256) transposed per window into vt[(w*256+c')*256+tok]
__global__ __launch_bounds__(256)
void gemm_wmma_kernel(const half_t* __restrict__ A, const half_t* __restrict__ Wt,
                      const float* __restrict__ bias, void* __restrict__ outp,
                      half_t* __restrict__ vt, int N, int K, int mode) {
  int lane = threadIdx.x & 31;
  int wid  = threadIdx.x >> 5;
  int row0 = blockIdx.x * 16;
  int col0 = (blockIdx.y * 8 + wid) * 16;
  int arow = row0 + (lane & 15);
  int brow = col0 + (lane & 15);
  int kshift = (lane & 16) ? 8 : 0;
  v8f acc = {};
  for (int k0 = 0; k0 < K; k0 += 32) {
    int kb = k0 + kshift;
    v8h alo = *(const v8h*)(A  + (size_t)arow * K + kb);
    v8h ahi = *(const v8h*)(A  + (size_t)arow * K + kb + 16);
    v8h blo = *(const v8h*)(Wt + (size_t)brow * K + kb);
    v8h bhi = *(const v8h*)(Wt + (size_t)brow * K + kb + 16);
    v16h a, b;
#pragma unroll
    for (int e = 0; e < 8; ++e) { a[e] = alo[e]; a[e + 8] = ahi[e]; b[e] = blo[e]; b[e + 8] = bhi[e]; }
    acc = __builtin_amdgcn_wmma_f32_16x16x32_f16(false, a, false, b, (short)0, acc, false, false);
  }
  int mbase = row0 + ((lane & 16) ? 8 : 0);
  int col = col0 + (lane & 15);
  float bv = bias[col];
#pragma unroll
  for (int r = 0; r < 8; ++r) {
    int m = mbase + r;
    size_t idx = (size_t)m * N + col;
    float v = acc[r] + bv;
    if (mode == 0)      ((float*)outp)[idx] = v;
    else if (mode == 1) ((float*)outp)[idx] += v;
    else if (mode == 2) ((half_t*)outp)[idx] = (half_t)v;
    else if (mode == 3) {
      float gv = 0.5f * v * (1.f + erff(v * 0.70710678118f));
      ((half_t*)outp)[idx] = (half_t)gv;
    } else {
      if (col < 256) ((half_t*)outp)[idx] = (half_t)v;
      else {
        int w = m >> 8, tok = m & 255;  // 256 tokens per window
        vt[((size_t)w * 256 + (col - 256)) * 256 + tok] = (half_t)v;
      }
    }
  }
}

// ---------- flash attention: one wave per (qtile, head, window) ----------
// Q rows: normal -> w*nq+qi ; cross -> gathered from x1 token map.
// K: pitch-512 token-major at col h*32. V: transposed buffers vt[(w*256+c)*256+tok].
__global__ __launch_bounds__(32)
void attn_wmma_kernel(const half_t* __restrict__ Q, const half_t* __restrict__ KVA,
                      const half_t* __restrict__ VTA, const half_t* __restrict__ KVB,
                      const half_t* __restrict__ VTB, half_t* __restrict__ Out,
                      int nq, int nkA, int nkB, int repB, int cross) {
  __shared__ half_t ptile[16 * 32];
  int lane = threadIdx.x;
  int qt = blockIdx.x, h = blockIdx.y, w = blockIdx.z;
  int lm = lane & 15;
  int kshift = (lane & 16) ? 8 : 0;

  int qi = qt * 16 + lm;
  size_t qrow;
  if (!cross) qrow = (size_t)w * nq + qi;
  else {
    int b = w >> 6, gy = (w >> 3) & 7, gx = w & 7, iy = qi >> 2, ix = qi & 3;
    int Y = gy * 4 + iy, X = gx * 4 + ix;
    int w1 = b * 4 + (Y >> 4) * 2 + (X >> 4);
    int t1 = (Y & 15) * 16 + (X & 15);
    qrow = (size_t)w1 * 256 + t1;
  }
  const half_t* qp = Q + qrow * 256 + h * 32;
  v8h qlo = *(const v8h*)(qp + kshift);
  v8h qhi = *(const v8h*)(qp + kshift + 16);
  v16h aq;
#pragma unroll
  for (int e = 0; e < 8; ++e) { aq[e] = qlo[e]; aq[e + 8] = qhi[e]; }

  float mr[8], lr[8];
  v8f olo = {}, ohi = {};
#pragma unroll
  for (int r = 0; r < 8; ++r) { mr[r] = -1e30f; lr[r] = 0.f; }
  const float scl = 0.1767766953f; // 1/sqrt(32)

  int nk = nkA + nkB;
  for (int ck = 0; ck < nk; ck += 32) {
    const half_t* kv; const half_t* vt; size_t wv; int t0;
    if (ck < nkA) { kv = KVA; vt = VTA; wv = (size_t)w; t0 = ck; }
    else          { kv = KVB; vt = VTB; wv = (size_t)(w / repB); t0 = ck - nkA; }
    size_t rb = wv * 256;

    // ---- S = Q * K^T (two 16-col tiles) ----
    const half_t* krl = kv + (rb + t0 + lm) * 512 + h * 32;
    const half_t* krh = kv + (rb + t0 + 16 + lm) * 512 + h * 32;
    v8h k1 = *(const v8h*)(krl + kshift), k2 = *(const v8h*)(krl + kshift + 16);
    v8h k3 = *(const v8h*)(krh + kshift), k4 = *(const v8h*)(krh + kshift + 16);
    v16h bkl, bkh;
#pragma unroll
    for (int e = 0; e < 8; ++e) { bkl[e] = k1[e]; bkl[e + 8] = k2[e]; bkh[e] = k3[e]; bkh[e + 8] = k4[e]; }
    v8f z = {};
    v8f slo = __builtin_amdgcn_wmma_f32_16x16x32_f16(false, aq, false, bkl, (short)0, z, false, false);
    v8f shi = __builtin_amdgcn_wmma_f32_16x16x32_f16(false, aq, false, bkh, (short)0, z, false, false);

    // ---- online softmax (rows live across 16-lane halves) ----
    float pl[8], ph[8];
#pragma unroll
    for (int r = 0; r < 8; ++r) {
      float sl = slo[r] * scl, sh = shi[r] * scl;
      float cm = fmaxf(sl, sh);
      for (int off = 1; off < 16; off <<= 1) cm = fmaxf(cm, __shfl_xor(cm, off));
      float mn = fmaxf(mr[r], cm);
      float al = __expf(mr[r] - mn);
      pl[r] = __expf(sl - mn);
      ph[r] = __expf(sh - mn);
      float s = pl[r] + ph[r];
      for (int off = 1; off < 16; off <<= 1) s += __shfl_xor(s, off);
      lr[r] = lr[r] * al + s;
      olo[r] *= al; ohi[r] *= al;
      mr[r] = mn;
    }

    // ---- re-layout P (C-layout f32 -> A-layout f16) via LDS ----
    __syncthreads();
    int mo = (lane & 16) ? 8 : 0;
#pragma unroll
    for (int r = 0; r < 8; ++r) {
      ptile[(r + mo) * 32 + lm]      = (half_t)pl[r];
      ptile[(r + mo) * 32 + 16 + lm] = (half_t)ph[r];
    }
    __syncthreads();
    v16h ap;
#pragma unroll
    for (int e = 0; e < 16; ++e) ap[e] = ptile[lm * 32 + kmap16(e, lane)];

    // ---- O += P * V  (V tiles: vectorized loads from transposed V) ----
    const half_t* vrl = vt + ((size_t)wv * 256 + h * 32 + lm) * 256 + t0;
    const half_t* vrh = vt + ((size_t)wv * 256 + h * 32 + 16 + lm) * 256 + t0;
    v8h v1 = *(const v8h*)(vrl + kshift), v2 = *(const v8h*)(vrl + kshift + 16);
    v8h v3 = *(const v8h*)(vrh + kshift), v4 = *(const v8h*)(vrh + kshift + 16);
    v16h bvl, bvh;
#pragma unroll
    for (int e = 0; e < 8; ++e) { bvl[e] = v1[e]; bvl[e + 8] = v2[e]; bvh[e] = v3[e]; bvh[e + 8] = v4[e]; }
    olo = __builtin_amdgcn_wmma_f32_16x16x32_f16(false, ap, false, bvl, (short)0, olo, false, false);
    ohi = __builtin_amdgcn_wmma_f32_16x16x32_f16(false, ap, false, bvh, (short)0, ohi, false, false);
    __syncthreads();
  }

  int mo = (lane & 16) ? 8 : 0;
#pragma unroll
  for (int r = 0; r < 8; ++r) {
    size_t orow = (size_t)w * nq + qt * 16 + r + mo;
    float inv = 1.f / lr[r];
    Out[orow * 256 + h * 32 + lm]      = (half_t)(olo[r] * inv);
    Out[orow * 256 + h * 32 + 16 + lm] = (half_t)(ohi[r] * inv);
  }
}

// ---------- scatter cross-attn projection back into x1 ----------
__global__ void cross_scatter_kernel(const float* __restrict__ xa, float* __restrict__ x1) {
  int r = blockIdx.x, c = threadIdx.x;        // r = wa*16+qi
  int wa = r >> 4, qi = r & 15;
  int b = wa >> 6, gy = (wa >> 3) & 7, gx = wa & 7, iy = qi >> 2, ix = qi & 3;
  int Y = gy * 4 + iy, X = gx * 4 + ix;
  int w1 = b * 4 + (Y >> 4) * 2 + (X >> 4);
  int t1 = (Y & 15) * 16 + (X & 15);
  x1[((size_t)w1 * 256 + t1) * 256 + c] += xa[(size_t)r * 256 + c];
}

// =================================================================
extern "C" void kernel_launch(void* const* d_in, const int* in_sizes, int n_in,
                              void* d_out, int out_size, void* d_ws, size_t ws_size,
                              hipStream_t stream) {
  // ---- inputs (setup_inputs dict order, nested dicts flattened) ----
  const float* scale0 = (const float*)d_in[0];
  const float* scale1 = (const float*)d_in[1];
  const float* pe0w1 = (const float*)d_in[2];
  const float* pe0b1 = (const float*)d_in[3];
  const float* pe0w2 = (const float*)d_in[4];
  const float* pe1w1 = (const float*)d_in[5];
  const float* pe1b1 = (const float*)d_in[6];
  const float* pe1w2 = (const float*)d_in[7];
  // blk0 @8, blk1 @22: ln1_w, ln1_b, wq, bq, wkv, bkv, wproj, bproj, ln2_w, ln2_b, w1m, b1m, w2m, b2m
  const float *b0[14], *b1[14];
  for (int i = 0; i < 14; ++i) { b0[i] = (const float*)d_in[8 + i]; b1[i] = (const float*)d_in[22 + i]; }

  // ---- workspace layout ----
  size_t off = 0;
  auto alloc = [&](size_t bytes) -> void* {
    void* p = (char*)d_ws + off;
    off += (bytes + 255) & ~(size_t)255;
    return p;
  };
  half_t *wt_q0 = (half_t*)alloc(256*256*2),  *wt_kv0 = (half_t*)alloc(512*256*2);
  half_t *wt_p0 = (half_t*)alloc(256*256*2),  *wt_m10 = (half_t*)alloc(1024*256*2);
  half_t *wt_m20 = (half_t*)alloc(256*1024*2);
  half_t *wt_q1 = (half_t*)alloc(256*256*2),  *wt_kv1 = (half_t*)alloc(512*256*2);
  half_t *wt_p1 = (half_t*)alloc(256*256*2),  *wt_m11 = (half_t*)alloc(1024*256*2);
  half_t *wt_m21 = (half_t*)alloc(256*1024*2);
  float  *tab0 = (float*)alloc(256*256*4),    *tab1 = (float*)alloc(256*256*4);
  float  *x0buf = (float*)alloc((size_t)R0*256*4);
  float  *x1buf = (float*)alloc((size_t)R1*256*4);
  half_t *lnb0  = (half_t*)alloc((size_t)R0*256*2);
  half_t *lnb1  = (half_t*)alloc((size_t)R1*256*2);
  half_t *q1b   = (half_t*)alloc((size_t)R1*256*2);
  half_t *kv1b  = (half_t*)alloc((size_t)R1*512*2);
  half_t *vt1b  = (half_t*)alloc((size_t)R1*256*2);
  half_t *kv1nb = (half_t*)alloc((size_t)R1*512*2);
  half_t *vt1nb = (half_t*)alloc((size_t)R1*256*2);
  half_t *ao1b  = (half_t*)alloc((size_t)R1*256*2);   // also reused as cross attn-out
  half_t *hid1b = (half_t*)alloc((size_t)R1*1024*2);
  float  *xab   = (float*)alloc((size_t)R1*256*4);
  half_t *vt0b  = (half_t*)alloc((size_t)R0*256*2);   // transposed V for scale0
  // big region: q0 | ao0 | kv0, overlaid by hid0 during the x0 MLP
  char* big = (char*)alloc((size_t)R0*1024*2);        // 64 MB
  half_t *q0b  = (half_t*)big;
  half_t *ao0b = (half_t*)(big + (size_t)R0*256*2);
  half_t *kv0b = (half_t*)(big + (size_t)R0*256*2*2);
  half_t *hid0b = (half_t*)big;
  (void)ws_size; (void)n_in; (void)in_sizes; (void)out_size;

  // ---- weight prep (transpose + f16) ----
  auto tc = [&](const float* W, half_t* Wt, int K, int N) {
    tcvt_kernel<<<(K*N)/256, 256, 0, stream>>>(W, Wt, K, N);
  };
  tc(b0[2], wt_q0, 256, 256);   tc(b0[4], wt_kv0, 256, 512);
  tc(b0[6], wt_p0, 256, 256);   tc(b0[10], wt_m10, 256, 1024);
  tc(b0[12], wt_m20, 1024, 256);
  tc(b1[2], wt_q1, 256, 256);   tc(b1[4], wt_kv1, 256, 512);
  tc(b1[6], wt_p1, 256, 256);   tc(b1[10], wt_m11, 256, 1024);
  tc(b1[12], wt_m21, 1024, 256);

  // ---- pos-embed + bottom-up ----
  pe_table_kernel<<<256, 256, 0, stream>>>(pe0w1, pe0b1, pe0w2, tab0);
  pe_table_kernel<<<256, 256, 0, stream>>>(pe1w1, pe1b1, pe1w2, tab1);
  add_pe_kernel<<<R0, 256, 0, stream>>>(scale0, tab0, x0buf);
  bottom_up_kernel<<<R1, 256, 0, stream>>>(scale1, tab1, x0buf, x1buf);

  auto gemm = [&](const half_t* A, const half_t* Wt, const float* bias, void* out,
                  half_t* vt, int rows, int N, int K, int mode) {
    gemm_wmma_kernel<<<dim3(rows/16, N/128), 256, 0, stream>>>(A, Wt, bias, out, vt, N, K, mode);
  };

  // ---- coarse scale self-attention + MLP ----
  ln_kernel<<<R1, 256, 0, stream>>>(x1buf, b1[0], b1[1], lnb1);
  gemm(lnb1, wt_q1, b1[3], q1b, nullptr, R1, 256, 256, 2);
  gemm(lnb1, wt_kv1, b1[5], kv1b, vt1b, R1, 512, 256, 4);
  attn_wmma_kernel<<<dim3(16, 8, 8), 32, 0, stream>>>(q1b, kv1b, vt1b, nullptr, nullptr,
                                                      ao1b, 256, 256, 0, 1, 0);
  gemm(ao1b, wt_p1, b1[7], x1buf, nullptr, R1, 256, 256, 1);
  ln_kernel<<<R1, 256, 0, stream>>>(x1buf, b1[8], b1[9], lnb1);
  gemm(lnb1, wt_m11, b1[11], hid1b, nullptr, R1, 1024, 256, 3);
  gemm(hid1b, wt_m21, b1[13], x1buf, nullptr, R1, 256, 1024, 1);

  // ---- fine scale: attend to own window + replicated coarse KV ----
  ln_kernel<<<R0, 256, 0, stream>>>(x0buf, b0[0], b0[1], lnb0);
  gemm(lnb0, wt_q0, b0[3], q0b, nullptr, R0, 256, 256, 2);
  gemm(lnb0, wt_kv0, b0[5], kv0b, vt0b, R0, 512, 256, 4);
  ln_kernel<<<R1, 256, 0, stream>>>(x1buf, b1[0], b1[1], lnb1);
  gemm(lnb1, wt_kv1, b1[5], kv1nb, vt1nb, R1, 512, 256, 4);
  attn_wmma_kernel<<<dim3(16, 8, 128), 32, 0, stream>>>(q0b, kv0b, vt0b, kv1nb, vt1nb,
                                                        ao0b, 256, 256, 256, 16, 0);
  gemm(ao0b, wt_p0, b0[7], x0buf, nullptr, R0, 256, 256, 1);
  ln_kernel<<<R0, 256, 0, stream>>>(x0buf, b0[8], b0[9], lnb0);
  gemm(lnb0, wt_m10, b0[11], hid0b, nullptr, R0, 1024, 256, 3);   // overlays q0/ao0/kv0 (dead)
  gemm(hid0b, wt_m20, b0[13], x0buf, nullptr, R0, 256, 1024, 1);

  // ---- one2one cross-attention aggregation ----
  ln_kernel<<<R1, 256, 0, stream>>>(x1buf, b1[0], b1[1], lnb1);
  gemm(lnb1, wt_q1, b1[3], q1b, nullptr, R1, 256, 256, 2);
  ln_kernel<<<R0, 256, 0, stream>>>(x0buf, b0[0], b0[1], lnb0);
  gemm(lnb0, wt_kv0, b0[5], kv0b, vt0b, R0, 512, 256, 4);         // fully rewritten
  attn_wmma_kernel<<<dim3(1, 8, 128), 32, 0, stream>>>(q1b, kv0b, vt0b, nullptr, nullptr,
                                                       ao1b, 16, 256, 0, 1, 1);
  gemm(ao1b, wt_p1, b1[7], xab, nullptr, R1, 256, 256, 0);
  cross_scatter_kernel<<<R1, 256, 0, stream>>>(xab, x1buf);
  ln_kernel<<<R1, 256, 0, stream>>>(x1buf, b1[8], b1[9], lnb1);
  gemm(lnb1, wt_m11, b1[11], hid1b, nullptr, R1, 1024, 256, 3);
  gemm(hid1b, wt_m21, b1[13], x1buf, nullptr, R1, 256, 1024, 1);

  // ---- emit (x0, x1) concatenated ----
  hipMemcpyAsync(d_out, x0buf, (size_t)R0 * 256 * 4, hipMemcpyDeviceToDevice, stream);
  hipMemcpyAsync((float*)d_out + (size_t)R0 * 256, x1buf, (size_t)R1 * 256 * 4,
                 hipMemcpyDeviceToDevice, stream);
}
